// mLSTMNetwork_9302899163222
// MI455X (gfx1250) — matrix operand order repaired and checked by reading, not compiled, not run on hardware
//
#include <hip/hip_runtime.h>

// ---------------------------------------------------------------------------
// mLSTM forward for MI455X (gfx1250).
// B=32, T=256, H=1024, L=4, O=256.  Weights prepacked fp32->bf16 into WMMA
// B-fragment order (50 MB, L2-resident).  Per (t,l): bf16 WMMA GEMM
// [32x1024]x[1024x6144] with split-K=4 (deterministic partial buffers),
// then a fused LayerNorm+gate+state kernel that also repacks h into the
// next layer's A fragments.
// ---------------------------------------------------------------------------

typedef __attribute__((ext_vector_type(16))) __bf16 v16bf;
typedef __attribute__((ext_vector_type(8)))  float  v8f;

#define HD    1024
#define BB    32
#define TT    256
#define LL    4
#define OO    256
#define NTI   384   // (6*1024)/16  N tiles
#define KTI   32    // 1024/32      total K tiles
#define SPLITK 4
#define KTS   (KTI / SPLITK)   // 8 K tiles per split

// round-to-nearest-even fp32 -> bf16
__device__ __forceinline__ __bf16 f2bf(float f) {
  unsigned u = __builtin_bit_cast(unsigned, f);
  unsigned r = u + 0x7FFFu + ((u >> 16) & 1u);
  unsigned short hv = (unsigned short)(r >> 16);
  return __builtin_bit_cast(__bf16, hv);
}

// A-fragment (16x32 bf16, M=batch) packed index for element (m, k):
// lanes 0-15 hold K chunks [0-7],[16-23]; lanes 16-31 hold [8-15],[24-31].
__device__ __forceinline__ int a_idx(int m, int k) {
  int mt = m >> 4, r = m & 15;
  int kt = k >> 5, kk = k & 31;
  int hi = kk >> 3;
  int lane = ((hi & 1) << 4) | r;
  int e    = ((hi >> 1) << 3) | (kk & 7);
  return (((mt * KTI + kt) * 32) + lane) * 16 + e;
}

// ---------------------------------------------------------------------------
// Prepack W[l][j][h][d] (fp32) -> Bpack bf16 in B-fragment (32x16 K x N) order:
// lane = ((kk>>3)&1)*16 + n%16 ; elem = (kk>>4)*8 + (kk&7).   Layout:
// Bpack[l][nt][kt][lane][e],  n = j*1024+d, k = h.
// ---------------------------------------------------------------------------
__global__ __launch_bounds__(256) void packw_kernel(
    const float* __restrict__ W, __bf16* __restrict__ bpack) {
  size_t tid = (size_t)blockIdx.x * 256 + threadIdx.x;   // < 4*6*1024*1024
  int d  = (int)(tid & 1023);
  size_t r = tid >> 10;
  int hh = (int)(r & 1023);  r >>= 10;
  int j  = (int)(r % 6);
  int l  = (int)(r / 6);
  float v = W[tid];
  int n  = j * 1024 + d;
  int nt = n >> 4, nn = n & 15;
  int kt = hh >> 5, kk = hh & 31;
  int lane = (((kk >> 3) & 1) << 4) | nn;
  int e    = ((kk >> 4) << 3) | (kk & 7);
  size_t idx = ((((size_t)l * NTI + nt) * KTI + kt) * 32 + lane) * 16 + e;
  bpack[idx] = f2bf(v);
}

// ---------------------------------------------------------------------------
// Pack x[:, t, :] into A fragments for the layer-0 GEMM.
// ---------------------------------------------------------------------------
__global__ __launch_bounds__(256) void packx_kernel(
    const float* __restrict__ x, __bf16* __restrict__ apack, int t) {
  int tid = blockIdx.x * 256 + threadIdx.x;   // < 32*1024
  int b = tid >> 10, d = tid & 1023;
  float v = x[(size_t)b * (TT * HD) + (size_t)t * HD + d];
  apack[a_idx(b, d)] = f2bf(v);
}

// ---------------------------------------------------------------------------
// GEMM: z[n][m] += A[m][k] * B[k][n], M=32, N=6144, K=1024, bf16 WMMA, f32 acc.
// Grid: (48, SPLITK) x 256 threads (8 waves).  Wave w of block bx handles
// N-tile nt = bx*8+w for both M-tiles over K-slice blockIdx.y.
// Partial sums go to zpart[split] (fp32 [6][32][1024]); summed in cell kernel.
// ---------------------------------------------------------------------------
__global__ __launch_bounds__(256) void gemm_kernel(
    const __bf16* __restrict__ apack,     // 2*32*32*16 bf16 = 64 KB
    const __bf16* __restrict__ bpack_l,   // this layer's packed weights
    float* __restrict__ zpart) {          // SPLITK partial z buffers
  __shared__ __bf16 As[2 * KTS * 32 * 16];   // this split's A slice: 16 KB

  const int tid   = threadIdx.x;
  const int split = blockIdx.y;

  // cooperative copy of this K-slice's A fragments to LDS.
  // Global A layout: [mt][kt][lane][e]; slice kt in [split*KTS, split*KTS+KTS).
  {
    const uint4* src0 = (const uint4*)(apack + (size_t)(0 * KTI + split * KTS) * 512);
    const uint4* src1 = (const uint4*)(apack + (size_t)(1 * KTI + split * KTS) * 512);
    uint4* dst0 = (uint4*)As;
    uint4* dst1 = (uint4*)(As + KTS * 512);
    // each half-slice = KTS*512 bf16 = 8192 B = 512 uint4 -> 2 per thread
#pragma unroll
    for (int i = 0; i < 2; ++i) {
      dst0[tid + 256 * i] = src0[tid + 256 * i];
      dst1[tid + 256 * i] = src1[tid + 256 * i];
    }
  }
  __syncthreads();

  const int wave = tid >> 5;
  const int lane = tid & 31;
  const int nt   = blockIdx.x * 8 + wave;          // 0..383

  v8f acc0 = {};
  v8f acc1 = {};

  const __bf16* bptr  = bpack_l + (size_t)nt * (KTI * 32 * 16)
                                + (size_t)(split * KTS) * 512 + lane * 16;
  const __bf16* aptr0 = As + lane * 16;                 // M-tile 0
  const __bf16* aptr1 = As + KTS * 512 + lane * 16;     // M-tile 1

#pragma unroll
  for (int kt = 0; kt < KTS; ++kt) {
    __builtin_prefetch(bptr + (kt + 2) * 512, 0, 3);  // WGP-scope prefetch
    v16bf bv = *(const v16bf*)(bptr  + kt * 512);
    v16bf a0 = *(const v16bf*)(aptr0 + kt * 512);
    v16bf a1 = *(const v16bf*)(aptr1 + kt * 512);
    acc0 = __builtin_amdgcn_wmma_f32_16x16x32_bf16(
        false, a0, false, bv, (short)0, acc0, false, false);
    acc1 = __builtin_amdgcn_wmma_f32_16x16x32_bf16(
        false, a1, false, bv, (short)0, acc1, false, false);
  }

  // C/D layout: lane holds n_local = lane&15; VGPR v holds m = (lane/16)*8 + v
  const int nglob = nt * 16 + (lane & 15);
  const int j = nglob >> 10;
  const int d = nglob & 1023;
  float* zrow = zpart + (size_t)split * (6 * BB * HD) + (size_t)j * (BB * HD) + d;
  const int mhalf = (lane >> 4) * 8;
#pragma unroll
  for (int vv = 0; vv < 8; ++vv) {
    zrow[(mhalf + vv) * HD]      = acc0[vv];   // M-tile 0 : batches 0..15
    zrow[(16 + mhalf + vv) * HD] = acc1[vv];   // M-tile 1 : batches 16..31
  }
}

// ---------------------------------------------------------------------------
// Fused cell: split-K sum + bias + 6x LayerNorm + gates + state update +
// repack h for the next layer's GEMM.  One block per batch element.
// ---------------------------------------------------------------------------
__device__ __forceinline__ float waveRed(float v) {
#pragma unroll
  for (int o = 16; o > 0; o >>= 1) v += __shfl_xor(v, o, 32);
  return v;
}

__global__ __launch_bounds__(1024) void cell_kernel(
    const float* __restrict__ zpart, // SPLITK x [6][32][1024]
    const float* __restrict__ bias,  // [6][1024]
    const float* __restrict__ g,     // [6][1024]
    const float* __restrict__ beta,  // [6][1024]
    float* __restrict__ cbuf,        // [32][1024] layer state (in d_out)
    float* __restrict__ nbuf,        // [32][1024] layer state (in d_out)
    float* __restrict__ hbuf,        // [32][1024] layer hidden (in d_out)
    float* __restrict__ act,         // [32][1024] fp32 activation buffer
    __bf16* __restrict__ apack,      // next-layer A fragments
    int t, int l) {
  __shared__ float sbuf[12 * 32];
  __shared__ float sout[12];

  const int b = blockIdx.x;
  const int d = threadIdx.x;
  const int w = d >> 5, ln = d & 31;

  float zv[6];
#pragma unroll
  for (int j = 0; j < 6; ++j) {
    float s = bias[j * HD + d];
#pragma unroll
    for (int sp = 0; sp < SPLITK; ++sp)
      s += zpart[(size_t)sp * (6 * BB * HD) + j * (BB * HD) + b * HD + d];
    zv[j] = s;
  }

  // --- 12 simultaneous block reductions: sum / sumsq per projection ---
  float part[12];
#pragma unroll
  for (int j = 0; j < 6; ++j) { part[2 * j] = zv[j]; part[2 * j + 1] = zv[j] * zv[j]; }
#pragma unroll
  for (int k = 0; k < 12; ++k) {
    float r = waveRed(part[k]);
    if (ln == 0) sbuf[k * 32 + w] = r;
  }
  __syncthreads();
  if (d < 12) {
    float s = 0.f;
#pragma unroll
    for (int i = 0; i < 32; ++i) s += sbuf[d * 32 + i];
    sout[d] = s;
  }
  __syncthreads();

  float lnv[6];
#pragma unroll
  for (int j = 0; j < 6; ++j) {
    float mu  = sout[2 * j] * (1.0f / HD);
    float var = sout[2 * j + 1] * (1.0f / HD) - mu * mu;
    float rs  = rsqrtf(var + 1e-5f);
    lnv[j] = (zv[j] - mu) * rs * g[j * HD + d] + beta[j * HD + d];
  }
  // order: q(unused), k, v, i, f, o
  const float kg = lnv[1];
  const float vg = lnv[2];
  const float ig = expf(lnv[3]);
  const float fg = 1.0f / (1.0f + expf(-lnv[4]));
  const float og = 1.0f / (1.0f + expf(-lnv[5]));

  // --- block reduce: sum(k) ---
  {
    float r = waveRed(kg);
    if (ln == 0) sbuf[w] = r;
  }
  __syncthreads();
  if (d == 0) {
    float s = 0.f;
#pragma unroll
    for (int i = 0; i < 32; ++i) s += sbuf[i];
    sout[0] = s;
  }
  __syncthreads();
  const float ksum = sout[0];

  const int idx = b * HD + d;
  const float cp = (t == 0) ? 0.f : cbuf[idx];
  const float np = (t == 0) ? 0.f : nbuf[idx];
  const float cc = fg * cp + ig * (ksum * vg);
  const float nn = fg * np + ig * kg;

  // --- block reduce: ||n||^2 ---
  {
    float r = waveRed(nn * nn);
    if (ln == 0) sbuf[w] = r;
  }
  __syncthreads();
  if (d == 0) {
    float s = 0.f;
#pragma unroll
    for (int i = 0; i < 32; ++i) s += sbuf[i];
    sout[0] = sqrtf(s);
  }
  __syncthreads();

  const float nh = nn / (sout[0] + 1e-8f);
  float hh = og * tanhf(cc * nh);
  if (l > 0) hh += act[idx];   // residual (act == this layer's input)

  cbuf[idx] = cc;
  nbuf[idx] = nn;
  hbuf[idx] = hh;
  act[idx]  = hh;              // fp32 input for next layer's residual
  if (l < 3) apack[a_idx(b, d)] = f2bf(hh);  // bf16 fragments for next GEMM
}

// ---------------------------------------------------------------------------
// Final FC: out[b][o] = h3[b] . fc_w[:,o] + fc_b[o]    (32x256, tiny)
// ---------------------------------------------------------------------------
__global__ __launch_bounds__(256) void fc_kernel(
    const float* __restrict__ h3, const float* __restrict__ fw,
    const float* __restrict__ fb, float* __restrict__ out) {
  const int b = blockIdx.x, o = threadIdx.x;
  const float* hr = h3 + b * HD;
  float acc = fb[o];
  for (int d = 0; d < HD; ++d) acc += hr[d] * fw[d * OO + o];
  out[b * OO + o] = acc;
}

// ---------------------------------------------------------------------------
extern "C" void kernel_launch(void* const* d_in, const int* in_sizes, int n_in,
                              void* d_out, int out_size, void* d_ws, size_t ws_size,
                              hipStream_t stream) {
  const float* x    = (const float*)d_in[0];   // [32,256,1024]
  const float* W    = (const float*)d_in[1];   // [4,6,1024,1024]
  const float* bias = (const float*)d_in[2];   // [4,6,1024]
  const float* ln_g = (const float*)d_in[3];   // [4,6,1024]
  const float* ln_b = (const float*)d_in[4];   // [4,6,1024]
  const float* fc_w = (const float*)d_in[5];   // [1024,256]
  const float* fc_b = (const float*)d_in[6];   // [256]

  // d_out = out[32*256] | h[4*32*1024] | c[...] | n[...]
  float* out  = (float*)d_out;
  float* hbuf = out  + BB * OO;
  float* cbuf = hbuf + LL * BB * HD;
  float* nbuf = cbuf + LL * BB * HD;

  // workspace: Bpack(50.3MB) | Apack(64KB) | zpart(4x768KB) | act(128KB)
  char* ws = (char*)d_ws;
  __bf16* Bpack = (__bf16*)ws;
  const size_t bpack_bytes = (size_t)LL * 6 * HD * HD * sizeof(__bf16);
  __bf16* Apack = (__bf16*)(ws + bpack_bytes);
  float*  zbuf  = (float*)(ws + bpack_bytes + 65536);
  float*  act   = (float*)(ws + bpack_bytes + 65536 +
                           (size_t)SPLITK * 6 * BB * HD * sizeof(float));

  // 1) prepack weights (once per call; L2-resident thereafter)
  packw_kernel<<<(LL * 6 * HD * HD) / 256, 256, 0, stream>>>(W, Bpack);

  // 2) sequential scan over time, layers pipelined via stream ordering
  for (int t = 0; t < TT; ++t) {
    packx_kernel<<<(BB * HD) / 256, 256, 0, stream>>>(x, Apack, t);
    for (int l = 0; l < LL; ++l) {
      gemm_kernel<<<dim3(48, SPLITK), 256, 0, stream>>>(
          Apack, Bpack + (size_t)l * 6 * HD * HD, zbuf);
      cell_kernel<<<BB, 1024, 0, stream>>>(
          zbuf,
          bias + l * 6 * HD, ln_g + l * 6 * HD, ln_b + l * 6 * HD,
          cbuf + l * BB * HD, nbuf + l * BB * HD, hbuf + l * BB * HD,
          act, Apack, t, l);
    }
  }

  // 3) output projection from top-layer hidden at final timestep
  fc_kernel<<<BB, 256, 0, stream>>>(hbuf + 3 * BB * HD, fc_w, fc_b, out);

  (void)in_sizes; (void)n_in; (void)out_size; (void)ws_size;
}
